// MultiheadAttention_36910948942228
// MI455X (gfx1250) — matrix operand order repaired and checked by reading, not compiled, not run on hardware
//
#include <hip/hip_runtime.h>
#include <hip/hip_bf16.h>

// ---------------------------------------------------------------------------
// MI455X (gfx1250) multi-head attention: f16 WMMA everywhere, flash softmax,
// async (ASYNCcnt-tracked) K/V tile staging into LDS with double buffering.
// B=2, S=4096, D=512, H=8, d_k=64.
// ---------------------------------------------------------------------------

typedef __attribute__((ext_vector_type(16))) _Float16 v16h;
typedef __attribute__((ext_vector_type(8)))  _Float16 v8h;
typedef __attribute__((ext_vector_type(4)))  _Float16 v4h;
typedef __attribute__((ext_vector_type(8)))  float    v8f;
typedef __attribute__((__vector_size__(4 * sizeof(int)))) int i32x4;
typedef __attribute__((address_space(1))) i32x4 gmem_i32x4;   // global int4
typedef __attribute__((address_space(3))) i32x4 lmem_i32x4;   // LDS int4

#define MHA_B 2
#define MHA_S 4096
#define MHA_D 512
#define MHA_H 8
#define MHA_DK 64
#define MHA_ROWS (MHA_B * MHA_S)   // 8192

__device__ __forceinline__ v8f wmma_f16(v16h a, v16h b, v8f c) {
  // D = A(16x32 f16) * B(32x16 f16) + C(16x16 f32)
  return __builtin_amdgcn_wmma_f32_16x16x32_f16(
      /*neg_a=*/false, a, /*neg_b=*/false, b,
      /*c_mod=*/(short)0, c, /*reuse_a=*/false, /*reuse_b=*/false);
}

// A-fragment (16x32, MxK): lane<16 holds K kk+0..15 of row lane,
// lane>=16 holds K kk+16..31 of row lane-16.
__device__ __forceinline__ v16h load_a_frag(const _Float16* rowbase, int kk, int lane) {
  const int o = kk + ((lane & 16) ? 16 : 0);
  v8h lo = *(const v8h*)(rowbase + o);
  v8h hi = *(const v8h*)(rowbase + o + 8);
  return __builtin_shufflevector(lo, hi, 0,1,2,3,4,5,6,7,8,9,10,11,12,13,14,15);
}

// B-fragment (32x16, KxN): lane<16 holds K kk+0..15 of column lane,
// lane>=16 holds K kk+16..31 of column lane-16 (contiguous 16 halves).
__device__ __forceinline__ v16h load_b_frag(const _Float16* colbase, int kk, int lane) {
  const int o = kk + ((lane & 16) ? 16 : 0);
  return *(const v16h*)(colbase + o);
}

// Async global -> LDS B128 copy pair (32 bytes per thread), ASYNCcnt-tracked.
__device__ __forceinline__ void async_copy_32B(const _Float16* g, _Float16* l) {
  gmem_i32x4* gp = (gmem_i32x4*)((__attribute__((address_space(1))) void*)(g));
  lmem_i32x4* lp = (lmem_i32x4*)((__attribute__((address_space(3))) void*)(l));
  __builtin_amdgcn_global_load_async_to_lds_b128(gp, lp, /*offset=*/0,  /*cpol=*/0);
  __builtin_amdgcn_global_load_async_to_lds_b128(gp, lp, /*offset=*/16, /*cpol=*/0);
}

// ---------------------------------------------------------------------------
// fp32 -> f16 conversion (n divisible by 4)
// ---------------------------------------------------------------------------
__global__ void cvt_f32_to_f16_kernel(const float* __restrict__ in,
                                      _Float16* __restrict__ out, int n) {
  int i = (blockIdx.x * blockDim.x + threadIdx.x) * 4;
  if (i + 3 < n) {
    float4 v = *(const float4*)(in + i);
    v4h o;
    o[0] = (_Float16)v.x; o[1] = (_Float16)v.y;
    o[2] = (_Float16)v.z; o[3] = (_Float16)v.w;
    *(v4h*)(out + i) = o;
  }
}

// ---------------------------------------------------------------------------
// Fused QKV projection:  out = X @ W^T + b,  X:[8192,512] f16, W:[512,512] f16.
// blockIdx.z selects q/k/v. Q,K written head-major [b,h,s,64]; V written
// transposed [b,h,64,s] so the P.V WMMA B-operand is a contiguous load.
// Block = 256 thr = 8 waves (4x2): block tile 128x128, wave tile 32x64.
// ---------------------------------------------------------------------------
__global__ __launch_bounds__(256) void qkv_proj_kernel(
    const _Float16* __restrict__ xq, const _Float16* __restrict__ xk,
    const _Float16* __restrict__ xv,
    const _Float16* __restrict__ wq, const _Float16* __restrict__ wk,
    const _Float16* __restrict__ wv,
    const float* __restrict__ bq, const float* __restrict__ bk,
    const float* __restrict__ bv,
    _Float16* __restrict__ Qh, _Float16* __restrict__ Kh,
    _Float16* __restrict__ Vt) {
  const int which = blockIdx.z;
  const _Float16* X = (which == 0) ? xq : (which == 1) ? xk : xv;
  const _Float16* W = (which == 0) ? wq : (which == 1) ? wk : wv;
  const float* bias = (which == 0) ? bq : (which == 1) ? bk : bv;

  const int lane = threadIdx.x & 31;
  const int wave = threadIdx.x >> 5;
  const int wm = wave & 3, wn = wave >> 2;
  const int m0 = blockIdx.y * 128 + wm * 32;  // rows of X (tokens)
  const int n0 = blockIdx.x * 128 + wn * 64;  // output features

  const _Float16* arow0 = X + (size_t)(m0 + (lane & 15)) * MHA_D;
  const _Float16* arow1 = arow0 + 16 * MHA_D;
  const _Float16* bcol[4];
#pragma unroll
  for (int ni = 0; ni < 4; ++ni)
    bcol[ni] = W + (size_t)(n0 + ni * 16 + (lane & 15)) * MHA_D;  // B[k][n]=W[n][k]

  v8f acc[2][4] = {};
  for (int kk = 0; kk < MHA_D; kk += 32) {
    v16h a0 = load_a_frag(arow0, kk, lane);
    v16h a1 = load_a_frag(arow1, kk, lane);
#pragma unroll
    for (int ni = 0; ni < 4; ++ni) {
      v16h bf = load_b_frag(bcol[ni], kk, lane);
      acc[0][ni] = wmma_f16(a0, bf, acc[0][ni]);
      acc[1][ni] = wmma_f16(a1, bf, acc[1][ni]);
    }
  }

  const int hi8 = (lane & 16) ? 8 : 0;
#pragma unroll
  for (int mi = 0; mi < 2; ++mi) {
#pragma unroll
    for (int ni = 0; ni < 4; ++ni) {
#pragma unroll
      for (int j = 0; j < 8; ++j) {
        const int row = m0 + mi * 16 + j + hi8;  // global token row 0..8191
        const int col = n0 + ni * 16 + (lane & 15);
        const float v = acc[mi][ni][j] + bias[col];
        const int bb = row >> 12, ss = row & (MHA_S - 1);
        const int hh = col >> 6, dd = col & 63;
        if (which < 2) {
          _Float16* dst = (which == 0) ? Qh : Kh;
          dst[(((size_t)(bb * MHA_H + hh)) * MHA_S + ss) * MHA_DK + dd] = (_Float16)v;
        } else {
          Vt[(((size_t)(bb * MHA_H + hh)) * MHA_DK + dd) * MHA_S + ss] = (_Float16)v;
        }
      }
    }
  }
}

// ---------------------------------------------------------------------------
// Flash attention. Grid (S/128, H, B). 8 waves x 16 query rows each.
// K/V tiles (64 x 64 f16) double-buffered in LDS via async global->LDS loads.
// Streaming softmax state (m,l) + per-iteration rescale factor in LDS.
// ---------------------------------------------------------------------------
__global__ __launch_bounds__(256) void flash_attn_kernel(
    const _Float16* __restrict__ Qh, const _Float16* __restrict__ Kh,
    const _Float16* __restrict__ Vt, _Float16* __restrict__ ctx) {
  __shared__ __align__(16) _Float16 sK[2][64][64];  // K tile [key][d], 16 KB
  __shared__ __align__(16) _Float16 sV[2][64][64];  // V^T tile [d][key], 16 KB
  __shared__ __align__(32) float    sS[8][16][64];  // raw scores (f32), 32 KB
  __shared__ __align__(32) _Float16 sP[8][16][64];  // probabilities, 16 KB
  __shared__ float sM[8][16];   // running row max
  __shared__ float sL[8][16];   // running row sum
  __shared__ float sR[8][16];   // per-iteration rescale

  const int tid  = threadIdx.x;
  const int lane = tid & 31;
  const int wave = tid >> 5;
  const int h = blockIdx.y, b = blockIdx.z;
  const size_t bh = (size_t)(b * MHA_H + h);
  const _Float16* Qb = Qh + bh * MHA_S * MHA_DK;
  const _Float16* Kb = Kh + bh * MHA_S * MHA_DK;
  const _Float16* Vb = Vt + bh * MHA_DK * MHA_S;

  const int q0 = blockIdx.x * 128 + wave * 16;  // this wave's 16 query rows

  // Per-thread async-copy coordinates: each thread moves 32B of K and 32B of V.
  const int cr = tid >> 2;        // row 0..63 (key row for K, d row for V)
  const int cc = (tid & 3) * 16;  // 0,16,32,48 (halves)

  // Preload Q A-fragments (16 x 64, two K-steps of 32)
  const _Float16* qrow = Qb + (size_t)(q0 + (lane & 15)) * MHA_DK;
  const v16h aq0 = load_a_frag(qrow, 0, lane);
  const v16h aq1 = load_a_frag(qrow, 32, lane);

  v8f o[4] = {};
  if (lane < 16) { sM[wave][lane] = -3.0e38f; sL[wave][lane] = 0.0f; }
  asm volatile("s_wait_dscnt 0x0" ::: "memory");

  // Kick off async staging of tile 0 into buffer 0.
  async_copy_32B(Kb + (size_t)cr * MHA_DK + cc, &sK[0][cr][cc]);
  async_copy_32B(Vb + (size_t)cr * MHA_S + cc,  &sV[0][cr][cc]);

  const float scale = 0.125f;  // 1/sqrt(64)
  const int hi8 = (lane & 16) ? 8 : 0;
  const int ntiles = MHA_S / 64;

  for (int it = 0; it < ntiles; ++it) {
    const int buf = it & 1;
    // Issue next tile (wraps at the end into a dead buffer; keeps the wait
    // immediate uniform and the schedule deterministic).
    {
      const int jn = ((it + 1) & (ntiles - 1)) * 64;
      const int nbuf = (it + 1) & 1;
      async_copy_32B(Kb + (size_t)(jn + cr) * MHA_DK + cc, &sK[nbuf][cr][cc]);
      async_copy_32B(Vb + (size_t)cr * MHA_S + jn + cc,    &sV[nbuf][cr][cc]);
    }
    if (it + 2 < ntiles) {  // warm L2 two tiles ahead
      const int j2 = (it + 2) * 64;
      __builtin_prefetch((const void*)(Kb + (size_t)j2 * MHA_DK), 0, 1);
      __builtin_prefetch((const void*)(Vb + j2), 0, 1);
    }
    // 4 newest async ops per wave (next tile) may be outstanding; current tile done.
    asm volatile("s_wait_asynccnt 0x4" ::: "memory");
    __syncthreads();

    // ---- S = (Q . K^T) * scale : 16 rows x 64 keys, 8 WMMAs (B from LDS) ----
    v8f sc[4];
#pragma unroll
    for (int ni = 0; ni < 4; ++ni) {
      const _Float16* kc = &sK[buf][ni * 16 + (lane & 15)][0];
      v8f c = {};
      c = wmma_f16(aq0, load_b_frag(kc, 0, lane), c);
      c = wmma_f16(aq1, load_b_frag(kc, 32, lane), c);
      sc[ni] = c;
    }
#pragma unroll
    for (int ni = 0; ni < 4; ++ni)
#pragma unroll
      for (int jr = 0; jr < 8; ++jr)
        sS[wave][jr + hi8][ni * 16 + (lane & 15)] = sc[ni][jr] * scale;
    asm volatile("s_wait_dscnt 0x0" ::: "memory");

    // ---- streaming softmax: each lane owns half a row (32 cols) ----
    {
      const int r = lane >> 1;
      const int c0 = (lane & 1) * 32;
      float mloc = -3.0e38f;
#pragma unroll
      for (int c = 0; c < 32; ++c) mloc = fmaxf(mloc, sS[wave][r][c0 + c]);
      mloc = fmaxf(mloc, __shfl_xor(mloc, 1, 32));
      const float mold = sM[wave][r];
      const float mnew = fmaxf(mold, mloc);
      const float resc = __expf(mold - mnew);
      float ssum = 0.0f;
#pragma unroll
      for (int c = 0; c < 32; ++c) {
        const float p = __expf(sS[wave][r][c0 + c] - mnew);
        sP[wave][r][c0 + c] = (_Float16)p;
        ssum += p;
      }
      ssum += __shfl_xor(ssum, 1, 32);
      if ((lane & 1) == 0) {
        sM[wave][r] = mnew;
        sL[wave][r] = sL[wave][r] * resc + ssum;
        sR[wave][r] = resc;
      }
    }
    asm volatile("s_wait_dscnt 0x0" ::: "memory");

    // ---- rescale running O by exp(m_old - m_new) ----
#pragma unroll
    for (int jr = 0; jr < 8; ++jr) {
      const float f = sR[wave][jr + hi8];
#pragma unroll
      for (int ni = 0; ni < 4; ++ni) o[ni][jr] *= f;
    }

    // ---- O += P . V : P as LDS A-fragment, V^T columns from LDS ----
    const _Float16* prow = &sP[wave][lane & 15][0];
    const v16h ap0 = load_a_frag(prow, 0, lane);
    const v16h ap1 = load_a_frag(prow, 32, lane);
#pragma unroll
    for (int ni = 0; ni < 4; ++ni) {
      const _Float16* vc = &sV[buf][ni * 16 + (lane & 15)][0];
      o[ni] = wmma_f16(ap0, load_b_frag(vc, 0, lane), o[ni]);
      o[ni] = wmma_f16(ap1, load_b_frag(vc, 32, lane), o[ni]);
    }
    __syncthreads();  // all waves done with buf before it is refilled
  }

  // ---- normalize and emit context, layout [B*S, 512] f16 ----
  float inv[8];
#pragma unroll
  for (int jr = 0; jr < 8; ++jr) inv[jr] = 1.0f / sL[wave][jr + hi8];
#pragma unroll
  for (int ni = 0; ni < 4; ++ni)
#pragma unroll
    for (int jr = 0; jr < 8; ++jr) {
      const int srow = q0 + jr + hi8;
      const size_t orow = (size_t)b * MHA_S + srow;
      ctx[orow * MHA_D + h * MHA_DK + ni * 16 + (lane & 15)] =
          (_Float16)(o[ni][jr] * inv[jr]);
    }
}

// ---------------------------------------------------------------------------
// Output projection: out = ctx @ w_o^T + b_o (fp32 out). Same tiling as QKV.
// ---------------------------------------------------------------------------
__global__ __launch_bounds__(256) void out_proj_kernel(
    const _Float16* __restrict__ ctx, const _Float16* __restrict__ wo,
    const float* __restrict__ bo, float* __restrict__ out) {
  const int lane = threadIdx.x & 31;
  const int wave = threadIdx.x >> 5;
  const int wm = wave & 3, wn = wave >> 2;
  const int m0 = blockIdx.y * 128 + wm * 32;
  const int n0 = blockIdx.x * 128 + wn * 64;

  const _Float16* arow0 = ctx + (size_t)(m0 + (lane & 15)) * MHA_D;
  const _Float16* arow1 = arow0 + 16 * MHA_D;
  const _Float16* bcol[4];
#pragma unroll
  for (int ni = 0; ni < 4; ++ni)
    bcol[ni] = wo + (size_t)(n0 + ni * 16 + (lane & 15)) * MHA_D;

  v8f acc[2][4] = {};
  for (int kk = 0; kk < MHA_D; kk += 32) {
    v16h a0 = load_a_frag(arow0, kk, lane);
    v16h a1 = load_a_frag(arow1, kk, lane);
#pragma unroll
    for (int ni = 0; ni < 4; ++ni) {
      v16h bf = load_b_frag(bcol[ni], kk, lane);
      acc[0][ni] = wmma_f16(a0, bf, acc[0][ni]);
      acc[1][ni] = wmma_f16(a1, bf, acc[1][ni]);
    }
  }

  const int hi8 = (lane & 16) ? 8 : 0;
#pragma unroll
  for (int mi = 0; mi < 2; ++mi)
#pragma unroll
    for (int ni = 0; ni < 4; ++ni)
#pragma unroll
      for (int j = 0; j < 8; ++j) {
        const int row = m0 + mi * 16 + j + hi8;
        const int col = n0 + ni * 16 + (lane & 15);
        out[(size_t)row * MHA_D + col] = acc[mi][ni][j] + bo[col];
      }
}

// ---------------------------------------------------------------------------
// Host launcher
// ---------------------------------------------------------------------------
extern "C" void kernel_launch(void* const* d_in, const int* in_sizes, int n_in,
                              void* d_out, int out_size, void* d_ws, size_t ws_size,
                              hipStream_t stream) {
  const float* query = (const float*)d_in[0];
  const float* key   = (const float*)d_in[1];
  const float* value = (const float*)d_in[2];
  const float* w_q   = (const float*)d_in[3];
  const float* b_q   = (const float*)d_in[4];
  const float* w_k   = (const float*)d_in[5];
  const float* b_k   = (const float*)d_in[6];
  const float* w_v   = (const float*)d_in[7];
  const float* b_v   = (const float*)d_in[8];
  const float* w_o   = (const float*)d_in[9];
  const float* b_o   = (const float*)d_in[10];
  float* out = (float*)d_out;

  const size_t XN = (size_t)MHA_ROWS * MHA_D;  // 4,194,304
  const size_t WN = (size_t)MHA_D * MHA_D;     //   262,144

  _Float16* p   = (_Float16*)d_ws;
  _Float16* xq  = p; p += XN;
  _Float16* xk  = p; p += XN;
  _Float16* xv  = p; p += XN;
  _Float16* wq16 = p; p += WN;
  _Float16* wk16 = p; p += WN;
  _Float16* wv16 = p; p += WN;
  _Float16* wo16 = p; p += WN;
  _Float16* Qh  = p; p += XN;   // [b,h,s,64]
  _Float16* Kh  = p; p += XN;   // [b,h,s,64]
  _Float16* Vt  = p; p += XN;   // [b,h,64,s]
  _Float16* ctx = p; p += XN;   // [b*s, 512]

  auto cvt = [&](const float* src, _Float16* dst, size_t n) {
    const int thr = 256;
    const int blocks = (int)((n / 4 + thr - 1) / thr);
    cvt_f32_to_f16_kernel<<<blocks, thr, 0, stream>>>(src, dst, (int)n);
  };
  cvt(query, xq, XN);
  cvt(key,   xk, XN);
  cvt(value, xv, XN);
  cvt(w_q, wq16, WN);
  cvt(w_k, wk16, WN);
  cvt(w_v, wv16, WN);
  cvt(w_o, wo16, WN);

  qkv_proj_kernel<<<dim3(MHA_D / 128, MHA_ROWS / 128, 3), 256, 0, stream>>>(
      xq, xk, xv, wq16, wk16, wv16, b_q, b_k, b_v, Qh, Kh, Vt);

  flash_attn_kernel<<<dim3(MHA_S / 128, MHA_H, MHA_B), 256, 0, stream>>>(
      Qh, Kh, Vt, ctx);

  out_proj_kernel<<<dim3(MHA_D / 128, MHA_ROWS / 128, 1), 256, 0, stream>>>(
      ctx, wo16, b_o, out);
}
